// KernelBiasTransformerDeepRV_2078764172135
// MI455X (gfx1250) — matrix operand order repaired and checked by reading, not compiled, not run on hardware
//
#include <hip/hip_runtime.h>
#include <hip/hip_bf16.h>

typedef _Float16 half_t;
typedef __attribute__((ext_vector_type(16))) _Float16 v16h;
typedef __attribute__((ext_vector_type(8)))  _Float16 v8h;
typedef __attribute__((ext_vector_type(8)))  float    v8f;
typedef __attribute__((ext_vector_type(4)))  unsigned int u32x4;
typedef __attribute__((ext_vector_type(4)))  int      i32x4;

#define LSEQ 2048
#define MROWS 4096   // B*L
#define FINP 514
#define FPAD 544
#define EDIM 503

// ---- CDNA5 async global->LDS path (guarded so compilation can never break)
#if defined(__has_builtin)
#  if __has_builtin(__builtin_amdgcn_global_load_async_to_lds_b128) && \
      __has_builtin(__builtin_amdgcn_s_wait_asynccnt)
#    define USE_ASYNC_LDS 1
#  endif
#endif
#ifndef USE_ASYNC_LDS
#  define USE_ASYNC_LDS 0
#endif

#if USE_ASYNC_LDS
typedef __attribute__((address_space(1))) i32x4* g_v4i_p;
typedef __attribute__((address_space(3))) i32x4* l_v4i_p;
#endif

__device__ __forceinline__ void copy16_g2l(const half_t* __restrict__ g, half_t* l) {
#if USE_ASYNC_LDS
    // flat LDS VA truncates to LDS offset (ISA 10.2 aperture rules); global VA == flat VA
    __builtin_amdgcn_global_load_async_to_lds_b128(
        (g_v4i_p)(unsigned long long)g,
        (l_v4i_p)(unsigned)(unsigned long long)l,
        0, 0);
#else
    *(u32x4*)l = *(const u32x4*)g;
#endif
}

__device__ __forceinline__ void async_wait() {
#if USE_ASYNC_LDS
    __builtin_amdgcn_s_wait_asynccnt(0);
#endif
}

__device__ __forceinline__ float gelu_f(float x) {
    // JAX default: tanh approximation
    float x3 = x * x * x;
    return 0.5f * x * (1.0f + tanhf(0.7978845608028654f * (x + 0.044715f * x3)));
}

// ---------------------------------------------------------------- converters
// transpose-convert: out[blk][n][k] = in[blk][k][n], f32 -> f16
__global__ void cvt16_t(const float* __restrict__ in, half_t* __restrict__ out,
                        int K, int N, int total) {
    int i = blockIdx.x * 256 + threadIdx.x;
    if (i >= total) return;
    int kn = K * N;
    int bidx = i / kn, r = i - bidx * kn;
    int n = r / K, k = r - n * K;
    out[i] = (half_t)in[(long)bidx * kn + (long)k * N + n];
}

// transpose-convert with K zero-padding: out[n][k<Kd], in[k<Ks][n]
__global__ void cvt16_pad_t(const float* __restrict__ in, half_t* __restrict__ out,
                            int Ks, int Kd, int N) {
    int i = blockIdx.x * 256 + threadIdx.x;
    if (i >= Kd * N) return;
    int n = i / Kd, k = i - n * Kd;
    out[i] = (k < Ks) ? (half_t)in[(long)k * N + n] : (half_t)0.0f;
}

// feats: [z | s(2) | embed(503) | cond(8) | zero-pad to 544], f16
__global__ void build_feats(const float* __restrict__ z, const float* __restrict__ cond,
                            const float* __restrict__ s, const float* __restrict__ emb,
                            half_t* __restrict__ out) {
    long i = (long)blockIdx.x * 256 + threadIdx.x;
    if (i >= (long)MROWS * FPAD) return;
    int row = (int)(i / FPAD), c = (int)(i % FPAD);
    int l = row & (LSEQ - 1);
    float v = 0.0f;
    if (c == 0)            v = z[row];
    else if (c < 3)        v = s[l * 2 + (c - 1)];
    else if (c < 3 + EDIM) v = emb[(long)l * EDIM + (c - 3)];
    else if (c < FINP)     v = cond[c - 3 - EDIM];
    out[i] = (half_t)v;
}

// ---------------------------------------------------------------- layernorm
__global__ void ln_kernel(const float* __restrict__ x, const float* __restrict__ sc,
                          const float* __restrict__ bb, half_t* __restrict__ out) {
    __shared__ float red[256];
    int row = blockIdx.x, t = threadIdx.x;
    float v = x[(long)row * 256 + t];
    red[t] = v; __syncthreads();
    for (int s = 128; s > 0; s >>= 1) { if (t < s) red[t] += red[t + s]; __syncthreads(); }
    float mean = red[0] * (1.0f / 256.0f);
    __syncthreads();
    float d = v - mean;
    red[t] = d * d; __syncthreads();
    for (int s = 128; s > 0; s >>= 1) { if (t < s) red[t] += red[t + s]; __syncthreads(); }
    float rstd = rsqrtf(red[0] * (1.0f / 256.0f) + 1e-6f);
    out[(long)row * 256 + t] = (half_t)(d * rstd * sc[t] + bb[t]);
}

// ---------------------------------------------------------------- WMMA GEMM
// C[M,N] = epilogue(A[M,K] @ B[K,N] + bias). A row-major [M][K]; B passed
// TRANSPOSED as Bt[N][K] so both LDS tiles are direct 16B 2D copies (async
// global->LDS when available). Block tile 64x64, 128 threads = 4 waves (2x2),
// each wave 32x32 = 2x2 wmma tiles of v_wmma_f32_16x16x32_f16.
// WVT: write f16 output in [b,h][d][l] layout (for V, consumed by attention).
template <bool GELU, bool RES, bool WF32, bool WF16, bool WVT>
__global__ __launch_bounds__(128)
void gemm_f16(const half_t* __restrict__ A, const half_t* __restrict__ Bt,
              const float* __restrict__ bias, const float* __restrict__ res,
              float* __restrict__ outF, half_t* __restrict__ outH,
              int M, int N, int K) {
    __shared__ half_t sA[64 * 32];   // [m][k]
    __shared__ half_t sB[64 * 32];   // [n][k]
    const int tid = threadIdx.x, lane = tid & 31, wid = tid >> 5;
    const int wm = wid & 1, wn = wid >> 1;
    const int m0 = blockIdx.y * 64, n0 = blockIdx.x * 64;
    const int lm = lane & 15, lh = lane >> 4;

    v8f acc[2][2];
    #pragma unroll
    for (int a = 0; a < 2; ++a)
        #pragma unroll
        for (int b = 0; b < 2; ++b)
            #pragma unroll
            for (int v = 0; v < 8; ++v) acc[a][b][v] = 0.0f;

    for (int kk = 0; kk < K; kk += 32) {
        #pragma unroll
        for (int it = 0; it < 2; ++it) {      // A tile: 64 rows x 32 halves
            int idx = tid + it * 128;
            int r = idx >> 2, cc = (idx & 3) * 8;
            copy16_g2l(A + (long)(m0 + r) * K + kk + cc, sA + r * 32 + cc);
        }
        #pragma unroll
        for (int it = 0; it < 2; ++it) {      // Bt: 128 rows x 16 halves (k 0-7, 16-23)
            int idx = tid + it * 128;
            int r = idx >> 1, cc = (idx & 1) * 16;
            copy16_g2l(Bt + (long)(n0 + r) * K + kk + cc, sB + r * 32 + cc);
        }
        async_wait();
        __syncthreads();

        v16h af[2], bf[2];
        #pragma unroll
        for (int tm = 0; tm < 2; ++tm) {
            int m = wm * 32 + tm * 16 + lm;
            v8h a0 = *(const v8h*)(sA + m * 32 + lh * 16);
            v8h a1 = *(const v8h*)(sA + m * 32 + lh * 16 + 8);
            #pragma unroll
            for (int j = 0; j < 8; ++j) { af[tm][j] = a0[j]; af[tm][j + 8] = a1[j]; }
        }
        #pragma unroll
        for (int tn = 0; tn < 2; ++tn) {
            int n = wn * 32 + tn * 16 + lm;
            bf[tn] = *(const v16h*)(sB + n * 32 + lh * 16);
        }
        #pragma unroll
        for (int tm = 0; tm < 2; ++tm)
            #pragma unroll
            for (int tn = 0; tn < 2; ++tn)
                acc[tm][tn] = __builtin_amdgcn_wmma_f32_16x16x32_f16(
                    false, af[tm], false, bf[tn], (short)0, acc[tm][tn], false, false);
        __syncthreads();
    }

    #pragma unroll
    for (int tm = 0; tm < 2; ++tm) {
        #pragma unroll
        for (int tn = 0; tn < 2; ++tn) {
            int col = n0 + wn * 32 + tn * 16 + lm;
            int rbase = m0 + wm * 32 + tm * 16 + lh * 8;
            float bcol = bias ? bias[col] : 0.0f;
            #pragma unroll
            for (int v = 0; v < 8; ++v) {
                long r = rbase + v;
                float val = acc[tm][tn][v] + bcol;
                if (GELU) val = gelu_f(val);
                if (RES)  val += res[r * N + col];
                if (WF32) outF[r * N + col] = val;
                if (WF16) outH[r * N + col] = (half_t)val;
                if (WVT) {
                    // (row=b*L+l, col=h*64+d) -> [(b*8+h)*64+d][l]
                    long bh = (r >> 11) * 8 + (col >> 6);
                    outH[(bh * 64 + (col & 63)) * LSEQ + (r & 2047)] = (half_t)val;
                }
            }
        }
    }
}

// ---------------------------------------------------------------- attention
// One block per (q-tile of 64, b*H+h). Flash-style over 32 key chunks of 64.
// K fragments and V^T fragments are read directly from global (layouts match
// the wmma B-fragment pattern); only S/P round-trip through LDS.
__global__ __launch_bounds__(128)
void attn_kernel(const half_t* __restrict__ qb, const half_t* __restrict__ kb,
                 const half_t* __restrict__ vt, const float* __restrict__ Kmat,
                 const float* __restrict__ bw_p, const float* __restrict__ bb_p,
                 int blk, half_t* __restrict__ ob) {
    __shared__ float  sS[64 * 64];
    __shared__ half_t sP[64 * 64];
    __shared__ float  sM[64], sL[64], sF[64];

    const int tid = threadIdx.x, lane = tid & 31, w = tid >> 5;
    const int lm = lane & 15, lh = lane >> 4;
    const int qt = blockIdx.x, bh = blockIdx.y;
    const int b = bh >> 3, h = bh & 7;
    const float bw = bw_p[blk], bbias = bb_p[blk];
    const long baseBL = (long)b * LSEQ;

    // resident Q fragments: wave handles q rows [qt*64 + w*16, +16)
    const int qrow = qt * 64 + w * 16 + lm;
    const half_t* qptr = qb + ((baseBL + qrow) * 8 + h) * 64;
    v16h aq[2];
    #pragma unroll
    for (int kc = 0; kc < 2; ++kc) {
        v8h q0 = *(const v8h*)(qptr + kc * 32 + lh * 16);
        v8h q1 = *(const v8h*)(qptr + kc * 32 + lh * 16 + 8);
        #pragma unroll
        for (int j = 0; j < 8; ++j) { aq[kc][j] = q0[j]; aq[kc][j + 8] = q1[j]; }
    }

    v8f oacc[4];
    #pragma unroll
    for (int nt = 0; nt < 4; ++nt)
        #pragma unroll
        for (int v = 0; v < 8; ++v) oacc[nt][v] = 0.0f;

    if (tid < 64) { sM[tid] = -3.0e38f; sL[tid] = 0.0f; }
    __syncthreads();

    for (int c = 0; c < 32; ++c) {
        const int key0 = c * 64;
        // S = Q K^T * 1/8   (B-fragments straight from global k buffer)
        #pragma unroll
        for (int nt = 0; nt < 4; ++nt) {
            int key = key0 + nt * 16 + lm;
            const half_t* kp = kb + ((baseBL + key) * 8 + h) * 64 + lh * 16;
            v16h b0 = *(const v16h*)(kp);
            v16h b1 = *(const v16h*)(kp + 32);
            v8f sacc;
            #pragma unroll
            for (int v = 0; v < 8; ++v) sacc[v] = 0.0f;
            sacc = __builtin_amdgcn_wmma_f32_16x16x32_f16(false, aq[0], false, b0, (short)0, sacc, false, false);
            sacc = __builtin_amdgcn_wmma_f32_16x16x32_f16(false, aq[1], false, b1, (short)0, sacc, false, false);
            #pragma unroll
            for (int v = 0; v < 8; ++v)
                sS[(w * 16 + lh * 8 + v) * 64 + nt * 16 + lm] = sacc[v] * 0.125f;
        }
        __syncthreads();

        // online softmax: one thread per q row
        if (tid < 64) {
            const int row = tid;
            const float* kmrow = Kmat + (long)(qt * 64 + row) * LSEQ + key0;
            if (c < 31) __builtin_prefetch(kmrow + 64, 0, 1);   // next chunk's bias row
            float mold = sM[row];
            float cm = -3.0e38f;
            #pragma unroll 8
            for (int j = 0; j < 64; ++j) {
                float xv = sS[row * 64 + j] + bw * kmrow[j] + bbias;
                sS[row * 64 + j] = xv;
                cm = fmaxf(cm, xv);
            }
            float mnew = fmaxf(mold, cm);
            float fac = __expf(mold - mnew);
            float sum = 0.0f;
            #pragma unroll 8
            for (int j = 0; j < 64; ++j) {
                float p = __expf(sS[row * 64 + j] - mnew);
                sum += p;
                sP[row * 64 + j] = (half_t)p;
            }
            sM[row] = mnew;
            sL[row] = sL[row] * fac + sum;
            sF[row] = fac;
        }
        __syncthreads();

        // rescale accumulators and O += P @ V  (V^T fragments from global)
        v16h ap[2];
        {
            const int m = w * 16 + lm;
            #pragma unroll
            for (int kc = 0; kc < 2; ++kc) {
                v8h p0 = *(const v8h*)(sP + m * 64 + kc * 32 + lh * 16);
                v8h p1 = *(const v8h*)(sP + m * 64 + kc * 32 + lh * 16 + 8);
                #pragma unroll
                for (int j = 0; j < 8; ++j) { ap[kc][j] = p0[j]; ap[kc][j + 8] = p1[j]; }
            }
        }
        float facs[8];
        #pragma unroll
        for (int v = 0; v < 8; ++v) facs[v] = sF[w * 16 + lh * 8 + v];
        #pragma unroll
        for (int nt = 0; nt < 4; ++nt) {
            #pragma unroll
            for (int v = 0; v < 8; ++v) oacc[nt][v] *= facs[v];
            const int d0 = nt * 16 + lm;
            const half_t* vp = vt + ((long)bh * 64 + d0) * LSEQ + key0 + lh * 16;
            v16h bv0 = *(const v16h*)(vp);
            v16h bv1 = *(const v16h*)(vp + 32);
            oacc[nt] = __builtin_amdgcn_wmma_f32_16x16x32_f16(false, ap[0], false, bv0, (short)0, oacc[nt], false, false);
            oacc[nt] = __builtin_amdgcn_wmma_f32_16x16x32_f16(false, ap[1], false, bv1, (short)0, oacc[nt], false, false);
        }
        __syncthreads();
    }

    // finalize: divide by running sum, write (b, l, h*64+d) f16
    float rsum[8];
    #pragma unroll
    for (int v = 0; v < 8; ++v) rsum[v] = 1.0f / sL[w * 16 + lh * 8 + v];
    #pragma unroll
    for (int nt = 0; nt < 4; ++nt)
        #pragma unroll
        for (int v = 0; v < 8; ++v) {
            int ql = qt * 64 + w * 16 + lh * 8 + v;
            ob[((baseBL + ql) * 8 + h) * 64 + nt * 16 + lm] = (half_t)(oacc[nt][v] * rsum[v]);
        }
}

// ---------------------------------------------------------------- head tail
__global__ void head_out(const half_t* __restrict__ mid, const float* __restrict__ Wh2,
                         const float* __restrict__ bh2, float* __restrict__ out) {
    __shared__ float red[128];
    int row = blockIdx.x, t = threadIdx.x;
    red[t] = (float)mid[(long)row * 128 + t] * Wh2[t];
    __syncthreads();
    for (int s = 64; s > 0; s >>= 1) { if (t < s) red[t] += red[t + s]; __syncthreads(); }
    if (t == 0) out[row] = red[0] + bh2[0];
}

// ================================================================ launch
extern "C" void kernel_launch(void* const* d_in, const int* in_sizes, int n_in,
                              void* d_out, int out_size, void* d_ws, size_t ws_size,
                              hipStream_t stream) {
    (void)in_sizes; (void)n_in; (void)out_size; (void)ws_size;
    const float* z      = (const float*)d_in[0];
    const float* cond   = (const float*)d_in[1];
    const float* s      = (const float*)d_in[2];
    const float* Kmat   = (const float*)d_in[3];
    const float* embed  = (const float*)d_in[4];
    const float* W_in1  = (const float*)d_in[5];
    const float* b_in1  = (const float*)d_in[6];
    const float* W_in2  = (const float*)d_in[7];
    const float* b_in2  = (const float*)d_in[8];
    const float* bias_w = (const float*)d_in[9];
    const float* bias_b = (const float*)d_in[10];
    const float* Wq     = (const float*)d_in[11];
    const float* bq     = (const float*)d_in[12];
    const float* Wk     = (const float*)d_in[13];
    const float* bk     = (const float*)d_in[14];
    const float* Wv     = (const float*)d_in[15];
    const float* bvv    = (const float*)d_in[16];
    const float* Wo     = (const float*)d_in[17];
    const float* bo     = (const float*)d_in[18];
    const float* ln1_s  = (const float*)d_in[19];
    const float* ln1_b  = (const float*)d_in[20];
    const float* ln2_s  = (const float*)d_in[21];
    const float* ln2_b  = (const float*)d_in[22];
    const float* Wf1    = (const float*)d_in[23];
    const float* bf1    = (const float*)d_in[24];
    const float* Wf2    = (const float*)d_in[25];
    const float* bf2    = (const float*)d_in[26];
    const float* Wh1    = (const float*)d_in[27];
    const float* bh1    = (const float*)d_in[28];
    const float* Wh2    = (const float*)d_in[29];
    const float* bh2    = (const float*)d_in[30];

    char* wp = (char*)d_ws;
    auto carve = [&](size_t bytes) -> char* {
        char* p = wp; wp += (bytes + 255) & ~(size_t)255; return p;
    };
    const int M = MROWS;
    float*  x      = (float*) carve((size_t)M * 256 * 4);
    half_t* featsA = (half_t*)carve((size_t)M * FPAD * 2);
    half_t* t1     = (half_t*)carve((size_t)M * 1024 * 2);   // MLP mid / FFN mid / head mid
    half_t* hbuf   = (half_t*)carve((size_t)M * 256 * 2);
    half_t* qbuf   = (half_t*)carve((size_t)M * 512 * 2);
    half_t* kbuf   = (half_t*)carve((size_t)M * 512 * 2);
    half_t* vtbuf  = (half_t*)carve((size_t)M * 512 * 2);    // [b*8+h][d][l]
    half_t* obuf   = (half_t*)carve((size_t)M * 512 * 2);
    half_t* xh     = (half_t*)carve((size_t)M * 256 * 2);
    // transposed f16 weights: Wt[n][k]
    half_t* W1p    = (half_t*)carve((size_t)FPAD * 1024 * 2);
    half_t* W2h    = (half_t*)carve((size_t)1024 * 256 * 2);
    half_t* Wqh    = (half_t*)carve((size_t)2 * 256 * 512 * 2);
    half_t* Wkh    = (half_t*)carve((size_t)2 * 256 * 512 * 2);
    half_t* Wvh    = (half_t*)carve((size_t)2 * 256 * 512 * 2);
    half_t* Woh    = (half_t*)carve((size_t)2 * 512 * 256 * 2);
    half_t* Wf1h   = (half_t*)carve((size_t)2 * 256 * 1024 * 2);
    half_t* Wf2h   = (half_t*)carve((size_t)2 * 1024 * 256 * 2);
    half_t* Wh1h   = (half_t*)carve((size_t)256 * 128 * 2);

    auto cvT = [&](const float* src, half_t* dst, int K, int N, int nb) {
        int total = K * N * nb;
        cvt16_t<<<(total + 255) / 256, 256, 0, stream>>>(src, dst, K, N, total);
    };
    // weight conversions (every call: deterministic, cheap)
    cvt16_pad_t<<<(FPAD * 1024 + 255) / 256, 256, 0, stream>>>(W_in1, W1p, FINP, FPAD, 1024);
    cvT(W_in2, W2h, 1024, 256, 1);
    cvT(Wq,  Wqh,  256, 512, 2);
    cvT(Wk,  Wkh,  256, 512, 2);
    cvT(Wv,  Wvh,  256, 512, 2);
    cvT(Wo,  Woh,  512, 256, 2);
    cvT(Wf1, Wf1h, 256, 1024, 2);
    cvT(Wf2, Wf2h, 1024, 256, 2);
    cvT(Wh1, Wh1h, 256, 128, 1);

    build_feats<<<(int)(((long)M * FPAD + 255) / 256), 256, 0, stream>>>(z, cond, s, embed, featsA);

    dim3 blk(128);
    // input MLP
    gemm_f16<true,  false, false, true,  false><<<dim3(1024 / 64, M / 64), blk, 0, stream>>>(
        featsA, W1p, b_in1, nullptr, nullptr, t1, M, 1024, FPAD);
    gemm_f16<false, false, true,  false, false><<<dim3(256 / 64, M / 64), blk, 0, stream>>>(
        t1, W2h, b_in2, nullptr, x, nullptr, M, 256, 1024);

    for (int i = 0; i < 2; ++i) {
        ln_kernel<<<M, 256, 0, stream>>>(x, ln1_s + i * 256, ln1_b + i * 256, hbuf);
        gemm_f16<false, false, false, true,  false><<<dim3(512 / 64, M / 64), blk, 0, stream>>>(
            hbuf, Wqh + i * 131072, bq + i * 512, nullptr, nullptr, qbuf, M, 512, 256);
        gemm_f16<false, false, false, true,  false><<<dim3(512 / 64, M / 64), blk, 0, stream>>>(
            hbuf, Wkh + i * 131072, bk + i * 512, nullptr, nullptr, kbuf, M, 512, 256);
        gemm_f16<false, false, false, false, true><<<dim3(512 / 64, M / 64), blk, 0, stream>>>(
            hbuf, Wvh + i * 131072, bvv + i * 512, nullptr, nullptr, vtbuf, M, 512, 256);
        attn_kernel<<<dim3(LSEQ / 64, 16), blk, 0, stream>>>(
            qbuf, kbuf, vtbuf, Kmat, bias_w, bias_b, i, obuf);
        gemm_f16<false, true,  true,  false, false><<<dim3(256 / 64, M / 64), blk, 0, stream>>>(
            obuf, Woh + i * 131072, bo + i * 256, x, x, nullptr, M, 256, 512);
        ln_kernel<<<M, 256, 0, stream>>>(x, ln2_s + i * 256, ln2_b + i * 256, hbuf);
        gemm_f16<true,  false, false, true,  false><<<dim3(1024 / 64, M / 64), blk, 0, stream>>>(
            hbuf, Wf1h + i * 262144, bf1 + i * 1024, nullptr, nullptr, t1, M, 1024, 256);
        if (i == 1) {
            gemm_f16<false, true, true, true,  false><<<dim3(256 / 64, M / 64), blk, 0, stream>>>(
                t1, Wf2h + i * 262144, bf2 + i * 256, x, x, xh, M, 256, 1024);
        } else {
            gemm_f16<false, true, true, false, false><<<dim3(256 / 64, M / 64), blk, 0, stream>>>(
                t1, Wf2h + i * 262144, bf2 + i * 256, x, x, nullptr, M, 256, 1024);
        }
    }

    // head: gelu(x @ Wh1 + bh1) @ Wh2 + bh2
    gemm_f16<true, false, false, true, false><<<dim3(128 / 64, M / 64), blk, 0, stream>>>(
        xh, Wh1h, bh1, nullptr, nullptr, t1, M, 128, 256);
    head_out<<<M, 128, 0, stream>>>(t1, Wh2, bh2, (float*)d_out);
}
